// EncDecTransformer_8959301780060
// MI455X (gfx1250) — matrix operand order repaired and checked by reading, not compiled, not run on hardware
//
#include <hip/hip_runtime.h>
#include <stdint.h>

// ---------------------------------------------------------------------------
// EncDec transformer forward for MI455X (gfx1250), wave32 + WMMA bf16.
// B=4, S=1024, D=768, H=12, HD=64, L=4, FF=3072, V=8000.
// All matmuls: V_WMMA_F32_16X16X32_BF16 (bf16 in, f32 accumulate).
// LDS double-buffered (ping-pong) with register prefetch; fragments loaded
// as aligned ds_load_b128 pairs (transposed B tiles, chunk-permuted A tiles).
// ---------------------------------------------------------------------------

typedef __attribute__((ext_vector_type(16))) __bf16 v16bf;
typedef __attribute__((ext_vector_type(8)))  __bf16 v8bf;
typedef __attribute__((ext_vector_type(8)))  float  v8f;

#define DEV static __device__ __forceinline__

DEV __bf16 f2bf(float f) {
  unsigned u = __builtin_bit_cast(unsigned, f);
  unsigned r = u + 0x7FFFu + ((u >> 16) & 1u);   // round-to-nearest-even
  unsigned short h = (unsigned short)(r >> 16);
  return __builtin_bit_cast(__bf16, h);
}

// Load a 32-byte fragment half-pair from LDS as two b128 loads.
DEV v16bf ldfrag(const __bf16* p) {
  p = (const __bf16*)__builtin_assume_aligned(p, 16);
  v8bf lo = *(const v8bf*)p;
  v8bf hi = *(const v8bf*)(p + 8);
  return __builtin_shufflevector(lo, hi, 0, 1, 2, 3, 4, 5, 6, 7,
                                 8, 9, 10, 11, 12, 13, 14, 15);
}

DEV v8f wmma_bf16(v16bf a, v16bf b, v8f c) {
  return __builtin_amdgcn_wmma_f32_16x16x32_bf16(false, a, false, b,
                                                 (short)0, c, false, false);
}

// A-operand K-chunk permutation: 8-elem chunk c (within a 32-wide K half)
// goes to position {0,2,1,3}[c] so per-lane fragment data is contiguous.
DEV int permchunk(int cb) { return (cb & ~3) | ((cb & 1) << 1) | ((cb >> 1) & 1); }
// Element-level version for 32-wide K half.
DEV int permelem(int c) { return (c & 7) | ((c & 16) >> 1) | ((c & 8) << 1); }

// ---------------------------------------------------------------------------
// fp32 -> bf16 bulk convert (weights)
// ---------------------------------------------------------------------------
__global__ __launch_bounds__(256) void cvt_bf16_kernel(const float* __restrict__ in,
                                                       __bf16* __restrict__ out,
                                                       size_t n) {
  size_t i = (size_t)blockIdx.x * 256u + threadIdx.x;
  if (i < n) out[i] = f2bf(in[i]);
}

// ---------------------------------------------------------------------------
// Embedding gather + positional add (fp32 out)
// ---------------------------------------------------------------------------
__global__ __launch_bounds__(256) void embed_kernel(const int* __restrict__ ids,
                                                    const float* __restrict__ emb,
                                                    const float* __restrict__ pos,
                                                    float* __restrict__ out,
                                                    int Dm, int Sl) {
  size_t tok = blockIdx.x;
  int id = ids[tok];
  int p  = (int)(tok % (size_t)Sl);
  const float* e  = emb + (size_t)id * Dm;
  const float* pp = pos + (size_t)p  * Dm;
  float* o = out + tok * (size_t)Dm;
  for (int i = threadIdx.x; i < Dm; i += 256) o[i] = e[i] + pp[i];
}

// ---------------------------------------------------------------------------
// LayerNorm: fp32 in -> bf16 out (GEMM operand). One token per workgroup.
// ---------------------------------------------------------------------------
__global__ __launch_bounds__(256) void layernorm_kernel(const float* __restrict__ x,
                                                        const float* __restrict__ g,
                                                        const float* __restrict__ bb,
                                                        __bf16* __restrict__ out,
                                                        int Dm) {
  __shared__ float red0[8];
  __shared__ float red1[8];
  int t = threadIdx.x;
  size_t row = blockIdx.x;
  const float* xr = x + row * (size_t)Dm;
  float s = 0.f, s2 = 0.f;
  for (int i = t; i < Dm; i += 256) { float v = xr[i]; s += v; s2 += v * v; }
  for (int d = 1; d < 32; d <<= 1) { s += __shfl_xor(s, d, 32); s2 += __shfl_xor(s2, d, 32); }
  if ((t & 31) == 0) { red0[t >> 5] = s; red1[t >> 5] = s2; }
  __syncthreads();
  if (t < 32) {
    float a  = (t < 8) ? red0[t] : 0.f;
    float a2 = (t < 8) ? red1[t] : 0.f;
    for (int d = 1; d < 8; d <<= 1) { a += __shfl_xor(a, d, 32); a2 += __shfl_xor(a2, d, 32); }
    if (t == 0) { red0[0] = a; red1[0] = a2; }
  }
  __syncthreads();
  float mean = red0[0] / (float)Dm;
  float var  = red1[0] / (float)Dm - mean * mean;
  float rstd = rsqrtf(var + 1e-5f);
  __bf16* o = out + row * (size_t)Dm;
  for (int i = t; i < Dm; i += 256) o[i] = f2bf((xr[i] - mean) * rstd * g[i] + bb[i]);
}

// ---------------------------------------------------------------------------
// Tiled WMMA bf16 GEMM: C[M,N] = A[M,K] x W[K,N] (+bias)(+residual)(+relu)
// 128 threads = 4 waves; block tile 128x64; wave tile 32x64 (8 WMMA accs).
// Double-buffered LDS with register prefetch: one barrier per K-step,
// global loads issued before the WMMA burst of the current step.
// As: chunk-permuted [m][kperm]; Bs: transposed [n][kperm]; b128 fragment loads.
// ---------------------------------------------------------------------------
__global__ __launch_bounds__(128) void gemm_bf16_kernel(
    const __bf16* __restrict__ A, const __bf16* __restrict__ W,
    const float* __restrict__ bias, const float* __restrict__ resid,
    float* __restrict__ outF, __bf16* __restrict__ outB,
    int M, int N, int K, int relu) {
  constexpr int AP = 40;
  constexpr int BP = 40;
  __shared__ __bf16 As[2][128 * AP];
  __shared__ __bf16 Bs[2][64 * BP];

  const int tid  = threadIdx.x;
  const int lane = tid & 31;
  const int w    = tid >> 5;
  const int nb   = N >> 6;
  const int bx   = blockIdx.x % nb;
  const int by   = blockIdx.x / nb;
  const int row0 = by << 7;
  const int col0 = bx << 6;
  const int g16  = (lane >> 4) << 4;

  auto ldAg = [&](int k0, v8bf* r) {   // 128 rows x 4 chunks -> 4 per thread
#pragma unroll
    for (int i = 0; i < 4; ++i) {
      const int ch = tid + (i << 7), rr = ch >> 2, cb = ch & 3;
      r[i] = *(const v8bf*)&A[(size_t)(row0 + rr) * K + k0 + (cb << 3)];
    }
  };
  auto stAl = [&](int buf, const v8bf* r) {
#pragma unroll
    for (int i = 0; i < 4; ++i) {
      const int ch = tid + (i << 7), rr = ch >> 2, cb = ch & 3;
      *(v8bf*)&As[buf][rr * AP + (permchunk(cb) << 3)] = r[i];
    }
  };
  auto ldBg = [&](int k0, v8bf* r) {   // 32 k-rows x 8 n-chunks -> 2 per thread
#pragma unroll
    for (int i = 0; i < 2; ++i) {
      const int ch = tid + (i << 7), rr = ch >> 3, cb = ch & 7;
      r[i] = *(const v8bf*)&W[(size_t)(k0 + rr) * N + col0 + (cb << 3)];
    }
  };
  auto stBl = [&](int buf, const v8bf* r) {  // transpose: Bs[n][kperm]
#pragma unroll
    for (int i = 0; i < 2; ++i) {
      const int ch = tid + (i << 7), rr = ch >> 3, cb = ch & 7;
#pragma unroll
      for (int j = 0; j < 8; ++j) Bs[buf][((cb << 3) + j) * BP + permelem(rr)] = r[i][j];
    }
  };

  v8f acc[2][4] = {};
  {  // prologue: stage K-slab 0 into buffer 0
    v8bf ra[4], rb[2];
    ldAg(0, ra); ldBg(0, rb);
    stAl(0, ra); stBl(0, rb);
  }
  __syncthreads();

  int cur = 0;
  for (int k0 = 0; k0 < K; k0 += 32) {
    const bool more = (k0 + 32) < K;
    v8bf ra[4], rb[2];
    if (more) { ldAg(k0 + 32, ra); ldBg(k0 + 32, rb); }  // prefetch to regs

    const int m0 = (w << 5) + (lane & 15);
    const v16bf a0 = ldfrag(&As[cur][m0 * AP + g16]);
    const v16bf a1 = ldfrag(&As[cur][(m0 + 16) * AP + g16]);
    v16bf bf[4];
#pragma unroll
    for (int t = 0; t < 4; ++t)
      bf[t] = ldfrag(&Bs[cur][((t << 4) + (lane & 15)) * BP + g16]);
#pragma unroll
    for (int t = 0; t < 4; ++t) {
      acc[0][t] = wmma_bf16(a0, bf[t], acc[0][t]);
      acc[1][t] = wmma_bf16(a1, bf[t], acc[1][t]);
    }

    if (more) { stAl(cur ^ 1, ra); stBl(cur ^ 1, rb); }
    __syncthreads();
    cur ^= 1;
  }

#pragma unroll
  for (int hlf = 0; hlf < 2; ++hlf)
#pragma unroll
    for (int t = 0; t < 4; ++t) {
      const int n = col0 + (t << 4) + (lane & 15);
      const float bv = bias ? bias[n] : 0.f;
#pragma unroll
      for (int r = 0; r < 8; ++r) {
        const int mr = row0 + (w << 5) + (hlf << 4) + r + ((lane >> 4) << 3);
        float v = acc[hlf][t][r] + bv;
        if (resid) v += resid[(size_t)mr * N + n];
        if (relu)  v = fmaxf(v, 0.f);
        if (outF)  outF[(size_t)mr * N + n] = v;
        if (outB)  outB[(size_t)mr * N + n] = f2bf(v);
      }
    }
}

// ---------------------------------------------------------------------------
// Flash attention, bf16 WMMA, wave32. One workgroup = one (b,h) x 64-query
// tile; 4 waves x 16 query rows; key blocks of 32, double-buffered K/V.
// Qs: chunk-permuted [q][dimperm]   (A operand of Q K^T)
// Ks: chunk-permuted [key][dimperm] (B operand of Q K^T)
// Vs: transposed     [dim][keyperm] (B operand of P V)
// Ps: elem-permuted  per-wave 16x32 (A operand of P V)
// ---------------------------------------------------------------------------
__global__ __launch_bounds__(128) void attn_kernel(
    const __bf16* __restrict__ Q, const __bf16* __restrict__ Kv,
    const __bf16* __restrict__ Vv, __bf16* __restrict__ O,
    int Hn, int Sq, int Sk, int Dm, int causal) {
  constexpr int QP = 72, KP = 72, VP = 40, PP = 40;
  __shared__ __bf16 Qs[64 * QP];
  __shared__ __bf16 Ks[2][32 * KP];
  __shared__ __bf16 Vs[2][64 * VP];
  __shared__ __bf16 Ps[4 * 16 * PP];

  const int tid  = threadIdx.x;
  const int lane = tid & 31;
  const int w    = tid >> 5;
  const int qt    = Sq >> 6;
  const int bh    = blockIdx.x / qt;
  const int qbase = (blockIdx.x % qt) << 6;
  const int b = bh / Hn, h = bh % Hn;
  const size_t qrow0 = (size_t)b * Sq;
  const size_t krow0 = (size_t)b * Sk;
  const int hoff = h << 6;
  const int g16  = (lane >> 4) << 4;

  auto ldKVg = [&](int kb, v8bf* rk, v8bf* rv) {  // 32 rows x 8 chunks
#pragma unroll
    for (int i = 0; i < 2; ++i) {
      const int ch = tid + (i << 7), rr = ch >> 3, cb = ch & 7;
      rk[i] = *(const v8bf*)&Kv[(krow0 + kb + rr) * Dm + hoff + (cb << 3)];
      rv[i] = *(const v8bf*)&Vv[(krow0 + kb + rr) * Dm + hoff + (cb << 3)];
    }
  };
  auto stKVl = [&](int buf, const v8bf* rk, const v8bf* rv) {
#pragma unroll
    for (int i = 0; i < 2; ++i) {
      const int ch = tid + (i << 7), rr = ch >> 3, cb = ch & 7;
      *(v8bf*)&Ks[buf][rr * KP + (permchunk(cb) << 3)] = rk[i];
#pragma unroll
      for (int j = 0; j < 8; ++j) Vs[buf][((cb << 3) + j) * VP + permelem(rr)] = rv[i][j];
    }
  };

  // stage Q (chunk-permuted within each 32-wide K half) + first K/V block
#pragma unroll
  for (int ch = tid; ch < 512; ch += 128) {
    const int r = ch >> 3, cb = ch & 7;
    v8bf v = *(const v8bf*)&Q[(qrow0 + qbase + r) * Dm + hoff + (cb << 3)];
    *(v8bf*)&Qs[r * QP + (permchunk(cb) << 3)] = v;
  }
  {
    v8bf rk[2], rv[2];
    ldKVg(0, rk, rv);
    stKVl(0, rk, rv);
  }
  __syncthreads();

  const int mloc = (w << 4) + (lane & 15);
  const v16bf aq0 = ldfrag(&Qs[mloc * QP + g16]);
  const v16bf aq1 = ldfrag(&Qs[mloc * QP + 32 + g16]);

  float mrow[8], lrow[8];
#pragma unroll
  for (int r = 0; r < 8; ++r) { mrow[r] = -1e30f; lrow[r] = 0.f; }
  v8f acc[4] = {};

  const int kend = causal ? (qbase + 64) : Sk;
  int cur = 0;
  for (int kb = 0; kb < kend; kb += 32) {
    const bool more = (kb + 32) < kend;
    v8bf rk[2], rv[2];
    if (more) ldKVg(kb + 32, rk, rv);  // prefetch next K/V block to regs

    // --- scores: S = (Q K^T) * 1/8, two 16-key subtiles ---
    v16bf bk[2][2];
#pragma unroll
    for (int nt = 0; nt < 2; ++nt) {
      const int n = (nt << 4) + (lane & 15);
      bk[nt][0] = ldfrag(&Ks[cur][n * KP + g16]);
      bk[nt][1] = ldfrag(&Ks[cur][n * KP + 32 + g16]);
    }
    v8f sc[2];
#pragma unroll
    for (int nt = 0; nt < 2; ++nt) {
      v8f z = {};
      z = wmma_bf16(aq0, bk[nt][0], z);
      z = wmma_bf16(aq1, bk[nt][1], z);
      const int cg = kb + (nt << 4) + (lane & 15);
#pragma unroll
      for (int r = 0; r < 8; ++r) {
        float s = z[r] * 0.125f;  // 1/sqrt(64)
        if (causal && cg > qbase + (w << 4) + r + ((lane >> 4) << 3)) s = -1e30f;
        z[r] = s;
      }
      sc[nt] = z;
    }

    // --- online softmax (row stats across the 16-lane column group) ---
    float alpha[8];
#pragma unroll
    for (int r = 0; r < 8; ++r) {
      float mx = fmaxf(sc[0][r], sc[1][r]);
      for (int d = 1; d < 16; d <<= 1) mx = fmaxf(mx, __shfl_xor(mx, d, 32));
      const float nm = fmaxf(mrow[r], mx);
      const float al = __expf(mrow[r] - nm);
      float p0 = __expf(sc[0][r] - nm);
      float p1 = __expf(sc[1][r] - nm);
      sc[0][r] = p0; sc[1][r] = p1;
      float rs = p0 + p1;
      for (int d = 1; d < 16; d <<= 1) rs += __shfl_xor(rs, d, 32);
      lrow[r] = lrow[r] * al + rs;
      mrow[r] = nm;
      alpha[r] = al;
    }
#pragma unroll
    for (int t = 0; t < 4; ++t)
#pragma unroll
      for (int r = 0; r < 8; ++r) acc[t][r] *= alpha[r];

    // --- repack P (C layout -> permuted A layout) via wave-private LDS ---
#pragma unroll
    for (int nt = 0; nt < 2; ++nt) {
      const int pc = permelem((nt << 4) + (lane & 15));
#pragma unroll
      for (int r = 0; r < 8; ++r) {
        const int mm = r + ((lane >> 4) << 3);
        Ps[((w << 4) + mm) * PP + pc] = f2bf(sc[nt][r]);
      }
    }
    // LDS ops are in-order within a wave: read back without barrier.
    const v16bf ap = ldfrag(&Ps[((w << 4) + (lane & 15)) * PP + g16]);

    // --- O += P V ---
    v16bf bv[4];
#pragma unroll
    for (int t = 0; t < 4; ++t)
      bv[t] = ldfrag(&Vs[cur][((t << 4) + (lane & 15)) * VP + g16]);
#pragma unroll
    for (int t = 0; t < 4; ++t) acc[t] = wmma_bf16(ap, bv[t], acc[t]);

    if (more) stKVl(cur ^ 1, rk, rv);
    __syncthreads();
    cur ^= 1;
  }

  // --- normalize and store O (bf16, [B*S, Dm] head-interleaved) ---
#pragma unroll
  for (int t = 0; t < 4; ++t) {
    const int n = hoff + (t << 4) + (lane & 15);
#pragma unroll
    for (int r = 0; r < 8; ++r) {
      const int mm = qbase + (w << 4) + r + ((lane >> 4) << 3);
      O[(qrow0 + mm) * Dm + n] = f2bf(acc[t][r] / lrow[r]);
    }
  }
}

// ---------------------------------------------------------------------------
// Host orchestration
// ---------------------------------------------------------------------------
extern "C" void kernel_launch(void* const* d_in, const int* in_sizes, int n_in,
                              void* d_out, int out_size, void* d_ws, size_t ws_size,
                              hipStream_t stream) {
  (void)in_sizes; (void)n_in; (void)out_size;
  constexpr int B = 4, S = 1024, D = 768, H = 12, L = 4, V = 8000, FF = 3072;
  constexpr int BS = B * S;

  const int*   src     = (const int*)  d_in[0];
  const int*   target  = (const int*)  d_in[2];
  const float* src_emb = (const float*)d_in[4];
  const float* tgt_emb = (const float*)d_in[5];
  const float* pos_emb = (const float*)d_in[6];
  const float* eaw  = (const float*)d_in[7];
  const float* eab  = (const float*)d_in[8];
  const float* elg  = (const float*)d_in[9];
  const float* elb  = (const float*)d_in[10];
  const float* ef1w = (const float*)d_in[11];
  const float* ef1b = (const float*)d_in[12];
  const float* ef2w = (const float*)d_in[13];
  const float* ef2b = (const float*)d_in[14];
  const float* eng  = (const float*)d_in[15];
  const float* enb  = (const float*)d_in[16];
  const float* daw  = (const float*)d_in[17];
  const float* dab  = (const float*)d_in[18];
  const float* dlg  = (const float*)d_in[19];
  const float* dlb  = (const float*)d_in[20];
  const float* df1w = (const float*)d_in[21];
  const float* df1b = (const float*)d_in[22];
  const float* df2w = (const float*)d_in[23];
  const float* df2b = (const float*)d_in[24];
  const float* dng  = (const float*)d_in[25];
  const float* dnb  = (const float*)d_in[26];
  const float* out_w = (const float*)d_in[27];
  const float* out_b = (const float*)d_in[28];
  float* logits = (float*)d_out;

  // ---- workspace carve-up ----
  char* p = (char*)d_ws;
  auto carve = [&](size_t bytes) -> char* {
    char* r = p; p += (bytes + 255) & ~(size_t)255; return r;
  };
  __bf16* wEncAtt = (__bf16*)carve((size_t)L * 4 * D * D * 2);
  __bf16* wDecAtt = (__bf16*)carve((size_t)L * 8 * D * D * 2);
  __bf16* wEncF1  = (__bf16*)carve((size_t)L * D * FF * 2);
  __bf16* wEncF2  = (__bf16*)carve((size_t)L * FF * D * 2);
  __bf16* wDecF1  = (__bf16*)carve((size_t)L * D * FF * 2);
  __bf16* wDecF2  = (__bf16*)carve((size_t)L * FF * D * 2);
  __bf16* wOut    = (__bf16*)carve((size_t)D * V * 2);
  float*  X       = (float*) carve((size_t)BS * D * 4);
  __bf16* XN      = (__bf16*)carve((size_t)BS * D * 2);
  __bf16* Qb      = (__bf16*)carve((size_t)BS * D * 2);
  __bf16* Kb      = (__bf16*)carve((size_t)BS * D * 2);
  __bf16* Vb      = (__bf16*)carve((size_t)BS * D * 2);
  __bf16* Ob      = (__bf16*)carve((size_t)BS * D * 2);
  __bf16* Hf      = (__bf16*)carve((size_t)BS * FF * 2);
  __bf16* MEMb    = (__bf16*)carve((size_t)BS * D * 2);
  if ((size_t)(p - (char*)d_ws) > ws_size) return;  // insufficient scratch

  auto cvt = [&](const float* in, __bf16* out, size_t n) {
    cvt_bf16_kernel<<<dim3((unsigned)((n + 255) / 256)), 256, 0, stream>>>(in, out, n);
  };
  auto gemm = [&](const __bf16* A, const __bf16* W, const float* bias,
                  const float* resid, float* oF, __bf16* oB,
                  int M, int N, int K, int relu) {
    gemm_bf16_kernel<<<dim3((unsigned)((M / 128) * (N / 64))), 128, 0, stream>>>(
        A, W, bias, resid, oF, oB, M, N, K, relu);
  };
  auto ln = [&](const float* x, const float* g, const float* bb, __bf16* out) {
    layernorm_kernel<<<dim3(BS), 256, 0, stream>>>(x, g, bb, out, D);
  };
  auto attn = [&](const __bf16* q, const __bf16* k, const __bf16* v, __bf16* o,
                  int causal) {
    attn_kernel<<<dim3(B * H * (S / 64)), 128, 0, stream>>>(q, k, v, o, H, S, S, D, causal);
  };

  // ---- weights to bf16 (once per launch) ----
  cvt(eaw,   wEncAtt, (size_t)L * 4 * D * D);
  cvt(daw,   wDecAtt, (size_t)L * 8 * D * D);
  cvt(ef1w,  wEncF1,  (size_t)L * D * FF);
  cvt(ef2w,  wEncF2,  (size_t)L * FF * D);
  cvt(df1w,  wDecF1,  (size_t)L * D * FF);
  cvt(df2w,  wDecF2,  (size_t)L * FF * D);
  cvt(out_w, wOut,    (size_t)D * V);

  const size_t DD = (size_t)D * D;
  const size_t DF = (size_t)D * FF;

  // ================= encoder =================
  embed_kernel<<<dim3(BS), 256, 0, stream>>>(src, src_emb, pos_emb, X, D, S);
  for (int l = 0; l < L; ++l) {
    const __bf16* Wl = wEncAtt + (size_t)l * 4 * DD;
    const float*  bl = eab + (size_t)l * 4 * D;
    ln(X, elg + (size_t)(l * 2 + 0) * D, elb + (size_t)(l * 2 + 0) * D, XN);
    gemm(XN, Wl + 0 * DD, bl + 0 * D, nullptr, nullptr, Qb, BS, D, D, 0);
    gemm(XN, Wl + 1 * DD, bl + 1 * D, nullptr, nullptr, Kb, BS, D, D, 0);
    gemm(XN, Wl + 2 * DD, bl + 2 * D, nullptr, nullptr, Vb, BS, D, D, 0);
    attn(Qb, Kb, Vb, Ob, /*causal=*/0);
    gemm(Ob, Wl + 3 * DD, bl + 3 * D, X, X, nullptr, BS, D, D, 0);
    ln(X, elg + (size_t)(l * 2 + 1) * D, elb + (size_t)(l * 2 + 1) * D, XN);
    gemm(XN, wEncF1 + (size_t)l * DF, ef1b + (size_t)l * FF, nullptr, nullptr, Hf,
         BS, FF, D, /*relu=*/1);
    gemm(Hf, wEncF2 + (size_t)l * DF, ef2b + (size_t)l * D, X, X, nullptr,
         BS, D, FF, 0);
  }
  ln(X, eng, enb, MEMb);  // encoder memory (bf16 operand for cross-attn K/V)

  // ================= decoder =================
  embed_kernel<<<dim3(BS), 256, 0, stream>>>(target, tgt_emb, pos_emb, X, D, S);
  for (int l = 0; l < L; ++l) {
    const __bf16* Wl = wDecAtt + (size_t)l * 8 * DD;
    const float*  bl = dab + (size_t)l * 8 * D;
    // self-attention (causal)
    ln(X, dlg + (size_t)(l * 3 + 0) * D, dlb + (size_t)(l * 3 + 0) * D, XN);
    gemm(XN, Wl + 0 * DD, bl + 0 * D, nullptr, nullptr, Qb, BS, D, D, 0);
    gemm(XN, Wl + 1 * DD, bl + 1 * D, nullptr, nullptr, Kb, BS, D, D, 0);
    gemm(XN, Wl + 2 * DD, bl + 2 * D, nullptr, nullptr, Vb, BS, D, D, 0);
    attn(Qb, Kb, Vb, Ob, /*causal=*/1);
    gemm(Ob, Wl + 3 * DD, bl + 3 * D, X, X, nullptr, BS, D, D, 0);
    // cross-attention (K/V from encoder memory)
    ln(X, dlg + (size_t)(l * 3 + 1) * D, dlb + (size_t)(l * 3 + 1) * D, XN);
    gemm(XN,   Wl + 4 * DD, bl + 4 * D, nullptr, nullptr, Qb, BS, D, D, 0);
    gemm(MEMb, Wl + 5 * DD, bl + 5 * D, nullptr, nullptr, Kb, BS, D, D, 0);
    gemm(MEMb, Wl + 6 * DD, bl + 6 * D, nullptr, nullptr, Vb, BS, D, D, 0);
    attn(Qb, Kb, Vb, Ob, /*causal=*/0);
    gemm(Ob, Wl + 7 * DD, bl + 7 * D, X, X, nullptr, BS, D, D, 0);
    // FFN
    ln(X, dlg + (size_t)(l * 3 + 2) * D, dlb + (size_t)(l * 3 + 2) * D, XN);
    gemm(XN, wDecF1 + (size_t)l * DF, df1b + (size_t)l * FF, nullptr, nullptr, Hf,
         BS, FF, D, /*relu=*/1);
    gemm(Hf, wDecF2 + (size_t)l * DF, df2b + (size_t)l * D, X, X, nullptr,
         BS, D, FF, 0);
  }
  ln(X, dng, dnb, XN);
  // logits [B*S, V], fp32 out
  gemm(XN, wOut, out_b, nullptr, logits, nullptr, BS, V, D, 0);
}
